// DynamicFC_88673894793270
// MI455X (gfx1250) — compile-verified, matrix operand order checked by reading, not run on hardware
//
#include <hip/hip_runtime.h>

typedef __attribute__((ext_vector_type(16))) __bf16 v16bf;
typedef __attribute__((ext_vector_type(8)))  float  v8f;
typedef __attribute__((ext_vector_type(8)))  unsigned int v8u;

#define CIN  512
#define COUT 512
#define HW   784
#define NB   64

// LDS layout (32-bit words). Fragments stored pre-swizzled into WMMA layout:
// word index = tile*256 + lane*8 + vgpr
#define A_HI 0        // 8 M-rows-of-16 tiles * 256 words
#define A_LO 2048
#define B_HI 4096     // 7 N-tiles * 256 words
#define B_LO 5888
#define LDS_WORDS 7680  // 30720 bytes, single-buffered (pipeline stages in VGPRs)

// Truncation split: hi = x with low 16 mantissa bits masked (x - hi is EXACT in
// fp32), lo = trunc16(x - hi). hi+lo captures 16 mantissa bits; the bf16x3
// product error stays ~2^-16 (dominated by the dropped lo*lo term).
// Packing two bf16 halves into one word = single v_perm_b32:
// dest = { u1.b3, u1.b2, u0.b3, u0.b2 }  -> selector 0x07060302 (S0=u1, S1=u0)
__device__ __forceinline__ void split2(float x0, float x1, unsigned &whi, unsigned &wlo) {
  unsigned u0 = __float_as_uint(x0), u1 = __float_as_uint(x1);
  float f0 = __uint_as_float(u0 & 0xFFFF0000u);
  float f1 = __uint_as_float(u1 & 0xFFFF0000u);
  whi = __builtin_amdgcn_perm(u1, u0, 0x07060302u);
  wlo = __builtin_amdgcn_perm(__float_as_uint(x1 - f1),
                              __float_as_uint(x0 - f0), 0x07060302u);
}

__device__ __forceinline__ v16bf ld_frag(const unsigned* p) {
  return __builtin_bit_cast(v16bf, *(const v8u*)p);   // 32B contiguous -> 2x ds_load_b128
}

// fragment slot for element with K index k (0..31), 16-bit data:
// lane group: k[3] selects lanes 0-15 vs 16-31; vgpr = (k&16)>>2 | (k&7)>>1; half = k&1
__device__ __forceinline__ int k_lane_grp(int k) { return ((k >> 3) & 1) << 4; }
__device__ __forceinline__ int k_vgpr(int k)     { return ((k & 16) >> 2) + ((k & 7) >> 1); }

__global__ __launch_bounds__(256)
void dynfc_bf16x3_kernel(const float* __restrict__ X,     // [B][Cin][HW]
                         const float* __restrict__ W,     // [B][Cout][Cin]
                         const float* __restrict__ bias,  // [B][Cout]
                         float* __restrict__ out)         // [B][Cout][HW]
{
  alignas(32) __shared__ unsigned lds[LDS_WORDS];

  const int tid = threadIdx.x;
  const int b   = blockIdx.z;
  const int m0  = blockIdx.y * 128;   // Cout tile
  const int n0  = blockIdx.x * 112;   // HW tile (7 * 112 = 784)

  const float* Xb = X + (size_t)b * CIN * HW;
  const float* Wb = W + (size_t)b * COUT * CIN + (size_t)m0 * CIN;

  const int wv   = tid >> 5;   // wave 0..7  -> M sub-row
  const int lane = tid & 31;

  v8f acc[7];
#pragma unroll
  for (int nt = 0; nt < 7; ++nt) acc[nt] = v8f{0.f,0.f,0.f,0.f,0.f,0.f,0.f,0.f};

  // ---- register staging buffers for the software pipeline ----
  float2 aReg[8];          // 128x32 A pairs: 2048 pairs / 256 thr = 8 each
  float  bR0[7], bR1[7];   // 32x112 B pairs: 1792 pairs / 256 thr = 7 each

  auto load_stage = [&](int ks) {
#pragma unroll
    for (int i = 0; i < 8; ++i) {
      int p = tid + i * 256;
      int m = p >> 4;                 // 0..127
      int k = (p & 15) << 1;          // even K in 0..30 (K contiguous in W)
      aReg[i] = *(const float2*)(Wb + (size_t)m * CIN + ks + k);
    }
#pragma unroll
    for (int i = 0; i < 7; ++i) {
      int p  = tid + i * 256;
      int kp = p / 112;
      int n  = p - kp * 112;          // consecutive threads -> consecutive n (coalesced)
      int k  = kp << 1;
      bR0[i] = Xb[(size_t)(ks + k)     * HW + n0 + n];
      bR1[i] = Xb[(size_t)(ks + k + 1) * HW + n0 + n];
    }
  };

  auto store_stage = [&]() {
#pragma unroll
    for (int i = 0; i < 8; ++i) {
      int p = tid + i * 256;
      int m = p >> 4;
      int k = (p & 15) << 1;
      unsigned whi, wlo; split2(aReg[i].x, aReg[i].y, whi, wlo);
      int idx = ((m >> 4) << 8) + (((m & 15) | k_lane_grp(k)) << 3) + k_vgpr(k);
      lds[A_HI + idx] = whi;
      lds[A_LO + idx] = wlo;
    }
#pragma unroll
    for (int i = 0; i < 7; ++i) {
      int p  = tid + i * 256;
      int kp = p / 112;
      int n  = p - kp * 112;
      int k  = kp << 1;
      unsigned whi, wlo; split2(bR0[i], bR1[i], whi, wlo);
      int idx = ((n >> 4) << 8) + (((n & 15) | k_lane_grp(k)) << 3) + k_vgpr(k);
      lds[B_HI + idx] = whi;
      lds[B_LO + idx] = wlo;
    }
  };

  // ---- pipeline prologue: stage K-step 0 into LDS ----
  load_stage(0);
  store_stage();

  for (int ks = 0; ks < CIN; ks += 32) {
    __syncthreads();                        // LDS stores of current stage visible
    if (ks + 32 < CIN) load_stage(ks + 32); // fire global loads for next stage now;
                                            // wait lands after the WMMA chain below

    const int ab = (wv << 8) + (lane << 3);
    v16bf ah = ld_frag(&lds[A_HI + ab]);
    v16bf al = ld_frag(&lds[A_LO + ab]);
#pragma unroll
    for (int nt = 0; nt < 7; ++nt) {
      const int bb = (nt << 8) + (lane << 3);
      v16bf bh = ld_frag(&lds[B_HI + bb]);
      v16bf bl = ld_frag(&lds[B_LO + bb]);
      // bf16x3: hi*hi + hi*lo + lo*hi  (~fp32 accuracy, drops 2^-16 lo*lo term)
      acc[nt] = __builtin_amdgcn_wmma_f32_16x16x32_bf16(false, ah, false, bh,
                                                        (short)0, acc[nt], false, false);
      acc[nt] = __builtin_amdgcn_wmma_f32_16x16x32_bf16(false, ah, false, bl,
                                                        (short)0, acc[nt], false, false);
      acc[nt] = __builtin_amdgcn_wmma_f32_16x16x32_bf16(false, al, false, bh,
                                                        (short)0, acc[nt], false, false);
    }

    __syncthreads();                        // all waves done reading LDS
    if (ks + 32 < CIN) store_stage();       // convert staged regs, write next stage
  }

  // ---- epilogue: bias + store (C/D layout: VGPR r, lane l -> M=r+8*(l/16), N=l%16) ----
  const int mbase = m0 + (wv << 4) + ((lane >> 4) << 3);
  const int nl    = lane & 15;
  const float* bp = bias + b * COUT;
  float bv[8];
#pragma unroll
  for (int r = 0; r < 8; ++r) bv[r] = bp[mbase + r];

  float* ob = out + (size_t)b * COUT * HW;
#pragma unroll
  for (int nt = 0; nt < 7; ++nt) {
    int n = n0 + (nt << 4) + nl;
#pragma unroll
    for (int r = 0; r < 8; ++r) {
      ob[(size_t)(mbase + r) * HW + n] = acc[nt][r] + bv[r];
    }
  }
}

extern "C" void kernel_launch(void* const* d_in, const int* in_sizes, int n_in,
                              void* d_out, int out_size, void* d_ws, size_t ws_size,
                              hipStream_t stream) {
  const float* X    = (const float*)d_in[0];   // input_  (B,Cin,H,W)
  const float* W    = (const float*)d_in[1];   // weight  (B,Cout,Cin,1,1)
  const float* bias = (const float*)d_in[2];   // bias    (B,Cout)
  float* out = (float*)d_out;                  // (B,Cout,H,W)

  dim3 grid(HW / 112, COUT / 128, NB);         // (7, 4, 64)
  dim3 block(256);
  hipLaunchKernelGGL(dynfc_bf16x3_kernel, grid, block, 0, stream, X, W, bias, out);
}
